// DON_cnn_79216376807825
// MI455X (gfx1250) — matrix operand order repaired and checked by reading, not compiled, not run on hardware
//
#include <hip/hip_runtime.h>
#include <hip/hip_bf16.h>

// ---------------------------------------------------------------------------
// DeepONet-style patch forward for MI455X (gfx1250, wave32, WMMA)
// Dominant cost: two 4-layer H=256 MLPs over 262144 points (~207 GFLOP,
// ~50k FLOP/byte -> pure matrix-throughput bound). f16 WMMA with f32
// accumulate, hardware V_TANH_F32 activations, weights pre-swizzled to the
// WMMA B-fragment layout in scratch (coalesced b128 loads).
// ---------------------------------------------------------------------------

#define H        256
#define TILE     128          // points per block in the WMMA MLP kernel
#define PAD      8            // f16 elements of row padding (LDS bank spread)
#define LSTR     (H + PAD)    // LDS row stride in f16 elements
#define MAXPATCH 8192
#define WMAT     (H * H)      // elements per 256x256 weight matrix

typedef _Float16 v16h __attribute__((ext_vector_type(16)));
typedef _Float16 v8h  __attribute__((ext_vector_type(8)));
typedef float    v8f  __attribute__((ext_vector_type(8)));

// gfx1250 hardware tanh (V_TANH_F32, TRANS32 pipe -> co-executes with WMMA).
__device__ __forceinline__ float fast_tanh(float v) {
#if __has_builtin(__builtin_amdgcn_tanhf)
  return __builtin_amdgcn_tanhf(v);
#elif __has_builtin(__builtin_amdgcn_tanh_f32)
  return __builtin_amdgcn_tanh_f32(v);
#else
  return tanhf(v);
#endif
}

// Monotonic float -> uint encoding so atomicMax(unsigned) implements float max.
__device__ __forceinline__ unsigned fkey(float f) {
  unsigned u = __float_as_uint(f);
  return (u & 0x80000000u) ? ~u : (u | 0x80000000u);
}
__device__ __forceinline__ float fdec(unsigned k) {
  unsigned u = (k & 0x80000000u) ? (k & 0x7FFFFFFFu) : ~k;
  return __uint_as_float(u);
}

// ---------------------------------------------------------------------------
// ws layout:
//   u32 [0..255]    tb (global_param) column-max keys
//   u32 [256..511]  br (local_param)  column-max keys
//   u32 [512]       MM (patch point count)
//   u32 [514..514+MAXPATCH) compacted patch indices
//   then (256B aligned): 6 x 65536 f16 packed weight matrices
//     (tb w1,w2,w3, br w1,w2,w3) in WMMA B-fragment order
// ---------------------------------------------------------------------------
__global__ __launch_bounds__(1024) void init_ws_kernel(unsigned* ws) {
  int t = threadIdx.x;
  if (t < 514) ws[t] = 0u;
}

// Pre-swizzle one 256x256 f32 weight matrix into f16 WMMA B-fragment order:
// packed[((n16*8 + ks)*32 + lane)*16 + j] = W[(ks*32 + (lane>>4)*16 + j)*H + n16*16 + (lane&15)]
__global__ __launch_bounds__(256) void pack_w_kernel(const float* __restrict__ W,
                                                     _Float16* __restrict__ P) {
  int idx = blockIdx.x * 256 + threadIdx.x;          // 0..65535 (grid 256)
  int j    = idx & 15;
  int lane = (idx >> 4) & 31;
  int ks   = (idx >> 9) & 7;
  int n16  = idx >> 12;
  int k = ks * 32 + (lane >> 4) * 16 + j;
  int n = n16 * 16 + (lane & 15);
  P[idx] = (_Float16)W[k * H + n];
}

// Deterministic, ordered compaction of points with patch id 995
// (cx=9, cy=9, cz=5). Single block; chunked ballot + prefix scan preserves
// ascending index order exactly like np.nonzero.
__global__ __launch_bounds__(1024) void patch_scan_kernel(const float* __restrict__ x,
                                                          unsigned* __restrict__ cnt,
                                                          int* __restrict__ idxl,
                                                          int n) {
  __shared__ unsigned wsum[32];
  __shared__ unsigned carry;
  int t = threadIdx.x;
  int lane = t & 31, wv = t >> 5;
  if (t == 0) carry = 0u;
  __syncthreads();
  for (int base = 0; base < n; base += 1024) {
    int i = base + t;
    bool pred = false;
    if (i < n) {
      float x0 = x[3 * i + 0], x1 = x[3 * i + 1], x2 = x[3 * i + 2];
      int c0 = min(9, max(0, (int)floorf(x0 * 10.0f)));
      int c1 = min(9, max(0, (int)floorf(x1 * 10.0f)));
      int c2 = min(9, max(0, (int)floorf(x2 * 10.0f)));
      pred = (c0 * 100 + c1 * 10 + c2) == 995;
    }
    unsigned mask = (unsigned)__ballot(pred);
    unsigned rank = __popc(mask & ((1u << lane) - 1u));
    if (lane == 0) wsum[wv] = (unsigned)__popc(mask);
    __syncthreads();
    unsigned woff = 0, total = 0;
    for (int w = 0; w < 32; ++w) {
      unsigned v = wsum[w];
      if (w < wv) woff += v;
      total += v;
    }
    if (pred) {
      unsigned pos = carry + woff + rank;
      if (pos < MAXPATCH) idxl[pos] = i;
    }
    __syncthreads();
    if (t == 0) carry += total;
    __syncthreads();
  }
  if (t == 0) *cnt = (carry < MAXPATCH) ? carry : MAXPATCH;
}

// Build a WMMA A fragment (16x32 f16 per-lane layout) from the LDS row m:
// halves 0-15 = K[k0+16*hi .. +15]
__device__ __forceinline__ v16h load_a(const _Float16* __restrict__ row, int k0, int hi) {
  v8h alo = *(const v8h*)(row + k0 + 16 * hi);
  v8h ahi = *(const v8h*)(row + k0 + 16 * hi + 8);
  v16h a;
#pragma unroll
  for (int j = 0; j < 8; ++j) { a[j] = alo[j]; a[8 + j] = ahi[j]; }
  return a;
}

// ---------------------------------------------------------------------------
// WMMA MLP + column max:  keys[n] = max over all points of MLP(x)[n]
// Block = 256 threads = 8 waves; TILE points per block; each wave owns 32
// output columns. Activations ping-pong in LDS as f16; B operands live in
// registers. Two M-tiles processed per inner step (4 WMMAs per DS wait).
// ---------------------------------------------------------------------------
__global__ __launch_bounds__(256) void mlp_max_kernel(
    const float* __restrict__ x,
    const float* __restrict__ w0, const float* __restrict__ b0,
    const float* __restrict__ w1, const float* __restrict__ b1,
    const float* __restrict__ w2, const float* __restrict__ b2,
    const float* __restrict__ w3, const float* __restrict__ b3,
    const _Float16* __restrict__ wpacked,   // 3*WMAT f16 in fragment order, or null
    unsigned* __restrict__ keys) {
  __shared__ __align__(16) _Float16 hbuf[2][TILE][LSTR];
  __shared__ float xs[TILE][3];

  int t = threadIdx.x;
  int base = blockIdx.x * TILE;

  for (int i = t; i < TILE * 3; i += 256) xs[i / 3][i % 3] = x[base * 3 + i];
  __syncthreads();

  // layer 0: 3 -> 256, tanh (thread t owns output feature t)
  {
    float wa = w0[t], wb = w0[H + t], wc = w0[2 * H + t], bb = b0[t];
    for (int m = 0; m < TILE; ++m) {
      float v = fast_tanh(xs[m][0] * wa + xs[m][1] * wb + xs[m][2] * wc + bb);
      hbuf[0][m][t] = (_Float16)v;
    }
  }
  __syncthreads();

  const int lane = t & 31;
  const int hi = lane >> 4;
  const int ln = lane & 15;
  const int wv = t >> 5;
  const int nbase = wv * 32;

  const float* Ws[3] = {w1, w2, w3};
  const float* Bs[3] = {b1, b2, b3};

  float cmax0 = -3.402823466e38f, cmax1 = -3.402823466e38f;
  int cur = 0;

  for (int layer = 0; layer < 3; ++layer) {
    const float* __restrict__ Bv = Bs[layer];

    // ---- preload B fragments for this wave's 32 columns --------------------
    v16h bf[2][8];
    if (wpacked) {
      // coalesced: fragment (n16, ks) is 32 lanes x 32B contiguous
      const v16h* src = (const v16h*)(wpacked + (size_t)layer * WMAT);
#pragma unroll
      for (int nt = 0; nt < 2; ++nt) {
        int n16 = (nbase >> 4) + nt;
#pragma unroll
        for (int ks = 0; ks < 8; ++ks)
          bf[nt][ks] = src[((n16 * 8 + ks) * 32) + lane];
      }
    } else {
      const float* __restrict__ W = Ws[layer];
#pragma unroll
      for (int nt = 0; nt < 2; ++nt) {
        int n = nbase + nt * 16 + ln;
#pragma unroll
        for (int ks = 0; ks < 8; ++ks) {
          int k0 = ks * 32 + hi * 16;
          v16h f;
#pragma unroll
          for (int j = 0; j < 16; ++j) f[j] = (_Float16)W[(k0 + j) * H + n];
          bf[nt][ks] = f;
        }
      }
    }
    float bias0 = Bv[nbase + ln];
    float bias1 = Bv[nbase + 16 + ln];

    // ---- 2 M-tiles per step: 4 WMMAs per A-load group ----------------------
    for (int mt = 0; mt < TILE / 16; mt += 2) {
      const _Float16* row0 = &hbuf[cur][mt * 16 + ln][0];
      const _Float16* row1 = &hbuf[cur][mt * 16 + 16 + ln][0];
      v8f c00 = {}, c01 = {}, c10 = {}, c11 = {};
#pragma unroll
      for (int ks = 0; ks < 8; ++ks) {
        int k0 = ks * 32;
        v16h a0 = load_a(row0, k0, hi);
        v16h a1 = load_a(row1, k0, hi);
        c00 = __builtin_amdgcn_wmma_f32_16x16x32_f16(false, a0, false, bf[0][ks],
                                                     (short)0, c00, false, false);
        c01 = __builtin_amdgcn_wmma_f32_16x16x32_f16(false, a0, false, bf[1][ks],
                                                     (short)0, c01, false, false);
        c10 = __builtin_amdgcn_wmma_f32_16x16x32_f16(false, a1, false, bf[0][ks],
                                                     (short)0, c10, false, false);
        c11 = __builtin_amdgcn_wmma_f32_16x16x32_f16(false, a1, false, bf[1][ks],
                                                     (short)0, c11, false, false);
      }
      if (layer < 2) {
        // D layout: element r sits at M = r + 8*hi, N = ln (per 16 col block)
#pragma unroll
        for (int r = 0; r < 8; ++r) {
          int mm0 = mt * 16 + r + 8 * hi;
          int mm1 = mm0 + 16;
          hbuf[cur ^ 1][mm0][nbase + ln]      = (_Float16)fast_tanh(c00[r] + bias0);
          hbuf[cur ^ 1][mm0][nbase + 16 + ln] = (_Float16)fast_tanh(c01[r] + bias1);
          hbuf[cur ^ 1][mm1][nbase + ln]      = (_Float16)fast_tanh(c10[r] + bias0);
          hbuf[cur ^ 1][mm1][nbase + 16 + ln] = (_Float16)fast_tanh(c11[r] + bias1);
        }
      } else {
#pragma unroll
        for (int r = 0; r < 8; ++r) {
          cmax0 = fmaxf(cmax0, fmaxf(c00[r], c10[r]) + bias0);
          cmax1 = fmaxf(cmax1, fmaxf(c01[r], c11[r]) + bias1);
        }
      }
    }
    cur ^= 1;
    __syncthreads();
  }

  cmax0 = fmaxf(cmax0, __shfl_xor(cmax0, 16, 32));
  cmax1 = fmaxf(cmax1, __shfl_xor(cmax1, 16, 32));
  if (hi == 0) {
    atomicMax(&keys[nbase + ln],      fkey(cmax0));
    atomicMax(&keys[nbase + 16 + ln], fkey(cmax1));
  }
}

// ---------------------------------------------------------------------------
// Trunk + head over the ~262 patch points (negligible cost).
// ---------------------------------------------------------------------------
__global__ __launch_bounds__(256) void patch_mlp_kernel(
    const float* __restrict__ x, const float* __restrict__ y,
    const float* __restrict__ trw0, const float* __restrict__ trb0,
    const float* __restrict__ trw1, const float* __restrict__ trb1,
    const float* __restrict__ trw2, const float* __restrict__ trb2,
    const float* __restrict__ trw3, const float* __restrict__ trb3,
    const float* __restrict__ ow0, const float* __restrict__ ob0,
    const float* __restrict__ ow1, const float* __restrict__ ob1,
    const float* __restrict__ ow2, const float* __restrict__ ob2,
    const unsigned* __restrict__ tbkeys, const unsigned* __restrict__ brkeys,
    const unsigned* __restrict__ cnt, const int* __restrict__ idxl,
    float* __restrict__ out) {
  __shared__ float feat[3 * H];
  __shared__ float bufA[H], bufB[H];
  __shared__ float red[256];
  int t = threadIdx.x;
  int MM = (int)*cnt;
  if (MM > MAXPATCH) MM = MAXPATCH;

  for (int i = blockIdx.x; i < MM; i += gridDim.x) {
    int p = idxl[i];
    float x0 = x[3 * p], x1 = x[3 * p + 1], x2 = x[3 * p + 2];

    bufA[t] = fast_tanh(x0 * trw0[t] + x1 * trw0[H + t] + x2 * trw0[2 * H + t] + trb0[t]);
    __syncthreads();
    { float s = trb1[t]; for (int k = 0; k < H; ++k) s += bufA[k] * trw1[k * H + t];
      bufB[t] = fast_tanh(s); }
    __syncthreads();
    { float s = trb2[t]; for (int k = 0; k < H; ++k) s += bufB[k] * trw2[k * H + t];
      bufA[t] = fast_tanh(s); }
    __syncthreads();
    { float s = trb3[t]; for (int k = 0; k < H; ++k) s += bufA[k] * trw3[k * H + t];
      feat[t] = s; }                    // local_coord (no tanh on last layer)
    feat[H + t]     = fdec(brkeys[t]);  // local_param
    feat[2 * H + t] = fdec(tbkeys[t]);  // global_param
    __syncthreads();
    { float s = ob0[t]; for (int k = 0; k < 3 * H; ++k) s += feat[k] * ow0[k * H + t];
      bufA[t] = fast_tanh(s); }
    __syncthreads();
    { float s = ob1[t]; for (int k = 0; k < H; ++k) s += bufA[k] * ow1[k * H + t];
      bufB[t] = fast_tanh(s); }
    __syncthreads();
    red[t] = bufB[t] * ow2[t];
    __syncthreads();
    for (int s = 128; s > 0; s >>= 1) {
      if (t < s) red[t] += red[t + s];
      __syncthreads();
    }
    if (t == 0) {
      out[i]      = red[0] + ob2[0];  // pred_patch
      out[MM + i] = y[p];             // gt_patch
    }
    __syncthreads();
  }
}

// ---------------------------------------------------------------------------
extern "C" void kernel_launch(void* const* d_in, const int* in_sizes, int n_in,
                              void* d_out, int out_size, void* d_ws, size_t ws_size,
                              hipStream_t stream) {
  (void)n_in; (void)out_size;
  const float* x = (const float*)d_in[0];
  const float* y = (const float*)d_in[1];
  // tb: 2..9, br: 10..17, tr: 18..25, o: 26..31
  const float* tb[8]; const float* br[8]; const float* tr[8]; const float* o[6];
  for (int i = 0; i < 8; ++i) tb[i] = (const float*)d_in[2 + i];
  for (int i = 0; i < 8; ++i) br[i] = (const float*)d_in[10 + i];
  for (int i = 0; i < 8; ++i) tr[i] = (const float*)d_in[18 + i];
  for (int i = 0; i < 6; ++i) o[i]  = (const float*)d_in[26 + i];

  unsigned* ws     = (unsigned*)d_ws;
  unsigned* tbkeys = ws;          // 256
  unsigned* brkeys = ws + 256;    // 256
  unsigned* cnt    = ws + 512;
  int*      idxl   = (int*)(ws + 514);

  // packed-weight region (256B aligned past the index list)
  size_t pack_off = (((size_t)(514 + MAXPATCH) * 4) + 255) & ~(size_t)255;
  size_t pack_bytes = (size_t)6 * WMAT * sizeof(_Float16);
  bool use_pack = ws_size >= pack_off + pack_bytes;
  _Float16* wp = use_pack ? (_Float16*)((char*)d_ws + pack_off) : (_Float16*)0;

  int n = in_sizes[0] / 3;        // 262144 points

  init_ws_kernel<<<1, 1024, 0, stream>>>(ws);
  patch_scan_kernel<<<1, 1024, 0, stream>>>(x, cnt, idxl, n);

  if (use_pack) {
    const float* mats[6] = {tb[2], tb[4], tb[6], br[2], br[4], br[6]};
    for (int m = 0; m < 6; ++m)
      pack_w_kernel<<<256, 256, 0, stream>>>(mats[m], wp + (size_t)m * WMAT);
  }

  dim3 grid(n / TILE);
  mlp_max_kernel<<<grid, 256, 0, stream>>>(x, tb[0], tb[1], tb[2], tb[3],
                                           tb[4], tb[5], tb[6], tb[7],
                                           wp, tbkeys);
  mlp_max_kernel<<<grid, 256, 0, stream>>>(x, br[0], br[1], br[2], br[3],
                                           br[4], br[5], br[6], br[7],
                                           use_pack ? wp + (size_t)3 * WMAT : (_Float16*)0,
                                           brkeys);

  patch_mlp_kernel<<<64, 256, 0, stream>>>(
      x, y,
      tr[0], tr[1], tr[2], tr[3], tr[4], tr[5], tr[6], tr[7],
      o[0], o[1], o[2], o[3], o[4], o[5],
      tbkeys, brkeys, cnt, idxl, (float*)d_out);
}
